// RecurrentFFN_39986145526207
// MI455X (gfx1250) — compile-verified, matrix operand order checked by reading, not run on hardware
//
#include <hip/hip_runtime.h>
#include <hip/hip_bf16.h>

#define B_SZ 1024
#define D_SZ 1024
#define H_SZ 4096
#define C_SZ (D_SZ + H_SZ)   // 5120

typedef __attribute__((ext_vector_type(16))) __bf16 v16bf;
typedef __attribute__((ext_vector_type(8)))  float  v8f;

union FragBF { uint4 u4[2]; v16bf v; };

// ---------------------------------------------------------------------------
// Pack kernels: build bf16 A matrices for the WMMA GEMMs
// ---------------------------------------------------------------------------
__global__ __launch_bounds__(256) void pack_combined(const float* __restrict__ x,
                                                     const float* __restrict__ h,
                                                     __bf16* __restrict__ out) {
    int i = blockIdx.x * 256 + threadIdx.x;        // [0, B*C)
    int b = i / C_SZ;
    int k = i - b * C_SZ;
    float v = (k < D_SZ) ? x[b * D_SZ + k] : h[b * H_SZ + (k - D_SZ)];
    out[i] = (__bf16)v;
}

__global__ __launch_bounds__(256) void pack_cand(const float* __restrict__ x,
                                                 const float* __restrict__ h,
                                                 const float* __restrict__ r,
                                                 __bf16* __restrict__ out) {
    int i = blockIdx.x * 256 + threadIdx.x;
    int b = i / C_SZ;
    int k = i - b * C_SZ;
    float v;
    if (k < D_SZ) v = x[b * D_SZ + k];
    else {
        int j = b * H_SZ + (k - D_SZ);
        v = r[j] * h[j];
    }
    out[i] = (__bf16)v;
}

__global__ __launch_bounds__(256) void pack_f32_bf16(const float* __restrict__ in,
                                                     __bf16* __restrict__ out) {
    int i = blockIdx.x * 256 + threadIdx.x;
    out[i] = (__bf16)in[i];
}

// ---------------------------------------------------------------------------
// Row LayerNorm (one block per row)
// ---------------------------------------------------------------------------
__global__ __launch_bounds__(256) void ln_rows(const float* __restrict__ in,
                                               float* __restrict__ out,
                                               const float* __restrict__ gamma,
                                               const float* __restrict__ beta,
                                               int W) {
    int row = blockIdx.x;
    int tid = threadIdx.x;
    const float* p = in + (size_t)row * W;
    float s = 0.f, s2 = 0.f;
    for (int i = tid; i < W; i += 256) { float v = p[i]; s += v; s2 += v * v; }
    __shared__ float sh_s[256], sh_s2[256];
    sh_s[tid] = s; sh_s2[tid] = s2;
    __syncthreads();
    for (int off = 128; off > 0; off >>= 1) {
        if (tid < off) { sh_s[tid] += sh_s[tid + off]; sh_s2[tid] += sh_s2[tid + off]; }
        __syncthreads();
    }
    float mean = sh_s[0] / (float)W;
    float var  = sh_s2[0] / (float)W - mean * mean;
    float inv  = rsqrtf(var + 1e-3f);
    for (int i = tid; i < W; i += 256)
        out[(size_t)row * W + i] = (p[i] - mean) * inv * gamma[i] + beta[i];
}

// ---------------------------------------------------------------------------
// WMMA GEMM. Block tile 128x64, BK=32 (one v_wmma_f32_16x16x32_bf16 K-step).
// 256 threads = 8 waves in a 4(M) x 2(N) grid; each wave owns 2x2 16x16 frags
// per output matrix. Register-staged software pipeline: stage k+1 global loads
// issue before the WMMAs of stage k; regs are committed to LDS at the top of
// the next iteration. fp32 weights convert to bf16 on the regs->LDS path and
// are stored transposed [n][k] so B fragments load with the same per-lane
// pattern as A fragments (ISA 16-bit A 16x32 layout). L2 prefetch runs two
// stages ahead on the weight streams (dominant HBM traffic).
//   MODE 0: dual GEMM, epilogue sigmoid -> (u, r)
//   MODE 1: dual GEMM, epilogue swiglu + h_pre = 2h + u*(swiglu - h)
//   MODE 2: single GEMM, epilogue plain (+bias)
// ---------------------------------------------------------------------------
template <int MODE>
__global__ __launch_bounds__(256) void gemm_wmma(
    const __bf16* __restrict__ A,     // [M x K] row-major (stride K)
    const float*  __restrict__ W1,    // [K x N] row-major (stride N)
    const float*  __restrict__ W2,    // dual modes only
    const float*  __restrict__ bias1,
    const float*  __restrict__ bias2,
    const float*  __restrict__ u_in,  // MODE 1
    const float*  __restrict__ h_in,  // MODE 1
    float* __restrict__ out1,
    float* __restrict__ out2,         // MODE 0
    int K, int N) {
    constexpr int BM = 128, BN = 64, BK = 32;
    constexpr int LDA = BK + 8;                    // bf16 elems/row, keeps 16B align
    __shared__ __bf16 As[BM * LDA];                // 10.0 KB
    __shared__ __bf16 Bs1[BN * LDA];               //  5.0 KB
    __shared__ __bf16 Bs2[(MODE == 2) ? 8 : BN * LDA];

    const int tid  = threadIdx.x;
    const int lane = tid & 31;
    const int wave = tid >> 5;
    const int wm   = wave >> 1;                    // 0..3
    const int wn   = wave & 1;                     // 0..1
    const int m0   = blockIdx.x * BM;
    const int n0   = blockIdx.y * BN;

    // Per-thread chunk coordinates (fixed across stages)
    const int arow0 = (tid + 0)   >> 2, akc0 = ((tid + 0)   & 3) << 3;
    const int arow1 = (tid + 256) >> 2, akc1 = ((tid + 256) & 3) << 3;
    const int wkk0  = (tid + 0)   >> 4, wnc0 = ((tid + 0)   & 15) << 2;
    const int wkk1  = (tid + 256) >> 4, wnc1 = ((tid + 256) & 15) << 2;

    v8f acc1[2][2] = {};
    v8f acc2[2][2] = {};

    const int lhalf = (lane & 16) ? 8 : 0;         // K sub-block per ISA layout
    const int lrow  = lane & 15;

    // Staging registers for the software pipeline
    uint4  ra[2];
    float4 rw1[2], rw2[2];

    auto load_stage = [&](int k0) {
        ra[0]  = *(const uint4*)(A + (size_t)(m0 + arow0) * K + k0 + akc0);
        ra[1]  = *(const uint4*)(A + (size_t)(m0 + arow1) * K + k0 + akc1);
        rw1[0] = *(const float4*)(W1 + (size_t)(k0 + wkk0) * N + n0 + wnc0);
        rw1[1] = *(const float4*)(W1 + (size_t)(k0 + wkk1) * N + n0 + wnc1);
        if (MODE != 2) {
            rw2[0] = *(const float4*)(W2 + (size_t)(k0 + wkk0) * N + n0 + wnc0);
            rw2[1] = *(const float4*)(W2 + (size_t)(k0 + wkk1) * N + n0 + wnc1);
        }
    };
    auto commit_stage = [&]() {
        *(uint4*)(&As[arow0 * LDA + akc0]) = ra[0];
        *(uint4*)(&As[arow1 * LDA + akc1]) = ra[1];
        Bs1[(wnc0 + 0) * LDA + wkk0] = (__bf16)rw1[0].x;
        Bs1[(wnc0 + 1) * LDA + wkk0] = (__bf16)rw1[0].y;
        Bs1[(wnc0 + 2) * LDA + wkk0] = (__bf16)rw1[0].z;
        Bs1[(wnc0 + 3) * LDA + wkk0] = (__bf16)rw1[0].w;
        Bs1[(wnc1 + 0) * LDA + wkk1] = (__bf16)rw1[1].x;
        Bs1[(wnc1 + 1) * LDA + wkk1] = (__bf16)rw1[1].y;
        Bs1[(wnc1 + 2) * LDA + wkk1] = (__bf16)rw1[1].z;
        Bs1[(wnc1 + 3) * LDA + wkk1] = (__bf16)rw1[1].w;
        if (MODE != 2) {
            Bs2[(wnc0 + 0) * LDA + wkk0] = (__bf16)rw2[0].x;
            Bs2[(wnc0 + 1) * LDA + wkk0] = (__bf16)rw2[0].y;
            Bs2[(wnc0 + 2) * LDA + wkk0] = (__bf16)rw2[0].z;
            Bs2[(wnc0 + 3) * LDA + wkk0] = (__bf16)rw2[0].w;
            Bs2[(wnc1 + 0) * LDA + wkk1] = (__bf16)rw2[1].x;
            Bs2[(wnc1 + 1) * LDA + wkk1] = (__bf16)rw2[1].y;
            Bs2[(wnc1 + 2) * LDA + wkk1] = (__bf16)rw2[1].z;
            Bs2[(wnc1 + 3) * LDA + wkk1] = (__bf16)rw2[1].w;
        }
    };

    load_stage(0);                                  // prologue
    for (int k0 = 0; k0 < K; k0 += BK) {
        commit_stage();
        __syncthreads();

        // Issue next stage's global loads before doing any math (latency
        // hidden under fragment loads + WMMAs), and L2-prefetch two ahead.
        if (k0 + BK < K) {
            load_stage(k0 + BK);
            if (k0 + 2 * BK < K) {
                __builtin_prefetch(W1 + (size_t)(k0 + 2 * BK + wkk0) * N + n0 + wnc0, 0, 1);
                if (MODE != 2)
                    __builtin_prefetch(W2 + (size_t)(k0 + 2 * BK + wkk0) * N + n0 + wnc0, 0, 1);
            }
        }

        FragBF a[2], b1f[2], b2f[2];
#pragma unroll
        for (int fm = 0; fm < 2; ++fm) {
            int r = wm * 32 + fm * 16 + lrow;
            a[fm].u4[0] = *(const uint4*)(&As[r * LDA + lhalf]);
            a[fm].u4[1] = *(const uint4*)(&As[r * LDA + lhalf + 16]);
        }
#pragma unroll
        for (int fn = 0; fn < 2; ++fn) {
            int nn = wn * 32 + fn * 16 + lrow;
            b1f[fn].u4[0] = *(const uint4*)(&Bs1[nn * LDA + lhalf]);
            b1f[fn].u4[1] = *(const uint4*)(&Bs1[nn * LDA + lhalf + 16]);
            if (MODE != 2) {
                b2f[fn].u4[0] = *(const uint4*)(&Bs2[nn * LDA + lhalf]);
                b2f[fn].u4[1] = *(const uint4*)(&Bs2[nn * LDA + lhalf + 16]);
            }
        }
#pragma unroll
        for (int fm = 0; fm < 2; ++fm)
#pragma unroll
            for (int fn = 0; fn < 2; ++fn) {
                acc1[fm][fn] = __builtin_amdgcn_wmma_f32_16x16x32_bf16(
                    false, a[fm].v, false, b1f[fn].v, (short)0, acc1[fm][fn], false, false);
                if (MODE != 2)
                    acc2[fm][fn] = __builtin_amdgcn_wmma_f32_16x16x32_bf16(
                        false, a[fm].v, false, b2f[fn].v, (short)0, acc2[fm][fn], false, false);
            }
        __syncthreads();
    }

    // Epilogue. C/D frag layout: lanes 0-15 -> N=lane, M=v; lanes 16-31 -> N=lane-16, M=v+8
    const int col_l = lane & 15;
    const int row_h = (lane >> 4) << 3;
#pragma unroll
    for (int fm = 0; fm < 2; ++fm)
#pragma unroll
        for (int fn = 0; fn < 2; ++fn) {
            int gcol = n0 + wn * 32 + fn * 16 + col_l;
            int grow0 = m0 + wm * 32 + fm * 16 + row_h;
            float bb1 = bias1[gcol];
            float bb2 = (MODE != 2) ? bias2[gcol] : 0.f;
#pragma unroll
            for (int v = 0; v < 8; ++v) {
                size_t idx = (size_t)(grow0 + v) * N + gcol;
                float p = acc1[fm][fn][v] + bb1;
                if (MODE == 0) {
                    float q = acc2[fm][fn][v] + bb2;
                    out1[idx] = 1.f / (1.f + __expf(-p));
                    out2[idx] = 1.f / (1.f + __expf(-q));
                } else if (MODE == 1) {
                    float q  = acc2[fm][fn][v] + bb2;
                    float sw = p * (1.f / (1.f + __expf(-p))) * q;   // silu(gate)*up
                    float uu = u_in[idx];
                    float hh = h_in[idx];
                    out1[idx] = 2.f * hh + uu * (sw - hh);           // (1-u)h + u*sw + h
                } else {
                    out1[idx] = p;
                }
            }
        }
}

// ---------------------------------------------------------------------------
extern "C" void kernel_launch(void* const* d_in, const int* in_sizes, int n_in,
                              void* d_out, int out_size, void* d_ws, size_t ws_size,
                              hipStream_t stream) {
    const float* x       = (const float*)d_in[0];
    const float* h       = (const float*)d_in[1];
    const float* W_u     = (const float*)d_in[2];
    const float* b_u     = (const float*)d_in[3];
    const float* W_r     = (const float*)d_in[4];
    const float* b_r     = (const float*)d_in[5];
    const float* W_g     = (const float*)d_in[6];
    const float* b_g     = (const float*)d_in[7];
    const float* W_up    = (const float*)d_in[8];
    const float* b_up    = (const float*)d_in[9];
    const float* W_d     = (const float*)d_in[10];
    const float* b_d     = (const float*)d_in[11];
    const float* gamma_h = (const float*)d_in[12];
    const float* beta_h  = (const float*)d_in[13];
    const float* gamma_o = (const float*)d_in[14];
    const float* beta_o  = (const float*)d_in[15];

    char* ws = (char*)d_ws;
    size_t off = 0;
    auto alloc = [&](size_t bytes) -> void* {
        void* p = ws + off;
        off = (off + bytes + 255) & ~(size_t)255;
        return p;
    };
    __bf16* comb_bf = (__bf16*)alloc((size_t)B_SZ * C_SZ * 2);
    __bf16* cand_bf = (__bf16*)alloc((size_t)B_SZ * C_SZ * 2);
    __bf16* newh_bf = (__bf16*)alloc((size_t)B_SZ * H_SZ * 2);
    float*  ws_u    = (float*)alloc((size_t)B_SZ * H_SZ * 4);
    float*  ws_r    = (float*)alloc((size_t)B_SZ * H_SZ * 4);
    float*  ws_hpre = (float*)alloc((size_t)B_SZ * H_SZ * 4);
    float*  ws_tmp  = (float*)alloc((size_t)B_SZ * D_SZ * 4);

    float* out_seg  = (float*)d_out;                    // [B, D]
    float* newh_seg = out_seg + (size_t)B_SZ * D_SZ;    // [B, H]

    dim3 blk(256);

    // 1) combined = [x, h] in bf16
    pack_combined<<<dim3((B_SZ * C_SZ) / 256), blk, 0, stream>>>(x, h, comb_bf);

    // 2) u = sigmoid(combined@W_u+b_u), r = sigmoid(combined@W_r+b_r)
    gemm_wmma<0><<<dim3(B_SZ / 128, H_SZ / 64), blk, 0, stream>>>(
        comb_bf, W_u, W_r, b_u, b_r, nullptr, nullptr, ws_u, ws_r, C_SZ, H_SZ);

    // 3) cand = [x, r*h] in bf16
    pack_cand<<<dim3((B_SZ * C_SZ) / 256), blk, 0, stream>>>(x, h, ws_r, cand_bf);

    // 4) gate/up dual GEMM, fused swiglu + h_pre
    gemm_wmma<1><<<dim3(B_SZ / 128, H_SZ / 64), blk, 0, stream>>>(
        cand_bf, W_g, W_up, b_g, b_up, ws_u, h, ws_hpre, nullptr, C_SZ, H_SZ);

    // 5) new_h = LN_H(h_pre) -> second output segment
    ln_rows<<<dim3(B_SZ), blk, 0, stream>>>(ws_hpre, newh_seg, gamma_h, beta_h, H_SZ);

    // 6) new_h -> bf16
    pack_f32_bf16<<<dim3((B_SZ * H_SZ) / 256), blk, 0, stream>>>(newh_seg, newh_bf);

    // 7) pre_out = new_h @ W_d + b_d
    gemm_wmma<2><<<dim3(B_SZ / 128, D_SZ / 64), blk, 0, stream>>>(
        newh_bf, W_d, nullptr, b_d, nullptr, nullptr, nullptr, ws_tmp, nullptr, H_SZ, D_SZ);

    // 8) out = LN_D(pre_out) -> first output segment
    ln_rows<<<dim3(B_SZ), blk, 0, stream>>>(ws_tmp, out_seg, gamma_o, beta_o, D_SZ);

    (void)in_sizes; (void)n_in; (void)out_size; (void)ws_size;
}
